// DenseLayer_65566970741075
// MI455X (gfx1250) — 8-bit matrix operand order repaired and checked by reading, compiled, not run on hardware
//
#include <hip/hip_runtime.h>
#include <hip/hip_bf16.h>
#include <math.h>


typedef int   v4i  __attribute__((ext_vector_type(4)));
typedef int   v16i __attribute__((ext_vector_type(16)));
typedef float v4f  __attribute__((ext_vector_type(4)));
typedef float v8f  __attribute__((ext_vector_type(8)));

union V16 {
    v16i v;
    v4i  q[4];
    unsigned long long d[8];
};

#define DIM_B   8192
#define DIM_K   4096
#define DIM_N   4096

#define LDS_STRIDE 144
#define TILE_BYTES (128 * LDS_STRIDE)
#define BUF_BYTES  (2 * TILE_BYTES)

__device__ __forceinline__ unsigned int bin_byte(float f) {
    return f < 0.0f ? 0xB8u : 0x38u;
}

__device__ __forceinline__ float hw_tanh(float x) {
#if __has_builtin(__builtin_amdgcn_tanhf)
    return __builtin_amdgcn_tanhf(x);
#else
    return tanhf(x);
#endif
}

__device__ __forceinline__ void async_b128(unsigned int lds_off,
                                           const unsigned char* gsrc) {
    asm volatile("global_load_async_to_lds_b128 %0, %1, off"
                 :: "v"(lds_off), "v"(gsrc)
                 : "memory");
}
__device__ __forceinline__ void wait_async0() {
    asm volatile("s_wait_asynccnt 0x0" ::: "memory");
}

__global__ __launch_bounds__(256) void binarize_x_kernel(
        const float* __restrict__ x, unsigned int* __restrict__ o) {
    size_t t = (size_t)blockIdx.x * blockDim.x + threadIdx.x;
    const v4f* xp = (const v4f*)x + t * 4;
    v4i w;
#pragma unroll
    for (int i = 0; i < 4; ++i) {
        v4f f = xp[i];
        unsigned int word;
        word  = bin_byte(f.x);
        word |= bin_byte(f.y) << 8;
        word |= bin_byte(f.z) << 16;
        word |= bin_byte(f.w) << 24;
        w[i] = (int)word;
    }
    *((v4i*)o + t) = w;
}

__global__ __launch_bounds__(256) void binarize_wT_kernel(
        const float* __restrict__ w, unsigned char* __restrict__ bt) {
    __shared__ unsigned char tile[64 * 68];
    const int k0 = blockIdx.y * 64;
    const int n0 = blockIdx.x * 64;
    const int tx = threadIdx.x & 63;
    const int ty = threadIdx.x >> 6;

#pragma unroll
    for (int i = 0; i < 16; ++i) {
        int kl = ty * 16 + i;
        float f = w[(size_t)(k0 + kl) * DIM_N + (n0 + tx)];
        tile[tx * 68 + kl] = (unsigned char)bin_byte(f);
    }
    __syncthreads();

    const int kw = (threadIdx.x & 15) * 4;
    const int nl = threadIdx.x >> 4;
#pragma unroll
    for (int p = 0; p < 4; ++p) {
        int n = nl + p * 16;
        unsigned int word = *(const unsigned int*)&tile[n * 68 + kw];
        *(unsigned int*)(bt + (size_t)(n0 + n) * DIM_K + k0 + kw) = word;
    }
}

__global__ __launch_bounds__(256) void bgemm_fp8_kernel(
        const unsigned char* __restrict__ A,
        const unsigned char* __restrict__ Bt,
        const float* __restrict__ bias,
        float* __restrict__ out) {
    __shared__ unsigned char smem[2 * BUF_BYTES];

    const int Mblk = blockIdx.y * 128;
    const int Nblk = blockIdx.x * 128;

    const int tid  = threadIdx.x;
    const int lane = tid & 31;
    const int wave = tid >> 5;
    const int wm   = wave & 3;
    const int wn   = wave >> 2;
    const int Am0  = wm * 32;
    const int Bn0  = wn * 64;
    const int lrow = lane & 15;
    const int half = lane >> 4;

    const unsigned int lds0 = (unsigned int)(size_t)(&smem[0]);

    const int crow = tid >> 3;
    const int cb   = (tid & 7) * 16;
    const unsigned char* gA0 = A  + (size_t)(Mblk + crow) * DIM_K + cb;
    const unsigned char* gB0 = Bt + (size_t)(Nblk + crow) * DIM_K + cb;
    const unsigned int   lA0 = lds0 + crow * LDS_STRIDE + cb;
    const unsigned int   lB0 = lds0 + TILE_BYTES + crow * LDS_STRIDE + cb;

    const unsigned int fA0 = lds0 + (Am0 + lrow) * LDS_STRIDE + half * 16;
    const unsigned int fB0 = lds0 + TILE_BYTES + (Bn0 + lrow) * LDS_STRIDE + half * 16;

    v8f acc00 = {}, acc01 = {}, acc02 = {}, acc03 = {};
    v8f acc10 = {}, acc11 = {}, acc12 = {}, acc13 = {};

#pragma unroll
    for (int r = 0; r < 4; ++r) {
        async_b128(lA0 + r * 32 * LDS_STRIDE, gA0 + (size_t)r * 32 * DIM_K);
        async_b128(lB0 + r * 32 * LDS_STRIDE, gB0 + (size_t)r * 32 * DIM_K);
    }
    wait_async0();
    __syncthreads();

    const int NKS = DIM_K / 128;
    for (int ks = 0; ks < NKS; ++ks) {
        const unsigned int bufo = (ks & 1) ? BUF_BYTES : 0;

        if (ks + 1 < NKS) {
            const unsigned int nbufo = (ks & 1) ? 0 : BUF_BYTES;
            const int kn = (ks + 1) * 128;
#pragma unroll
            for (int r = 0; r < 4; ++r) {
                async_b128(lA0 + nbufo + r * 32 * LDS_STRIDE,
                           gA0 + kn + (size_t)r * 32 * DIM_K);
                async_b128(lB0 + nbufo + r * 32 * LDS_STRIDE,
                           gB0 + kn + (size_t)r * 32 * DIM_K);
            }
        }

        V16 b0, b1, b2, b3;
#pragma unroll
        for (int i = 0; i < 4; ++i) {
            b0.q[i] = *(const v4i*)&smem[(fB0 + bufo +  0 * LDS_STRIDE) - lds0 + i * 32];
            b1.q[i] = *(const v4i*)&smem[(fB0 + bufo + 16 * LDS_STRIDE) - lds0 + i * 32];
            b2.q[i] = *(const v4i*)&smem[(fB0 + bufo + 32 * LDS_STRIDE) - lds0 + i * 32];
            b3.q[i] = *(const v4i*)&smem[(fB0 + bufo + 48 * LDS_STRIDE) - lds0 + i * 32];
        }
        V16 a0, a1;
#pragma unroll
        for (int i = 0; i < 8; ++i) {
            a0.d[i] = *(const unsigned long long*)
                      &smem[(fA0 + bufo +  0 * LDS_STRIDE) - lds0 + (i >> 1) * 32 + (i & 1) * 8];
            a1.d[i] = *(const unsigned long long*)
                      &smem[(fA0 + bufo + 16 * LDS_STRIDE) - lds0 + (i >> 1) * 32 + (i & 1) * 8];
        }

        acc00 = __builtin_amdgcn_wmma_f32_16x16x128_fp8_fp8(a0.v, b0.v, (short)0, acc00, false, false);
        acc01 = __builtin_amdgcn_wmma_f32_16x16x128_fp8_fp8(a0.v, b1.v, (short)0, acc01, false, false);
        acc02 = __builtin_amdgcn_wmma_f32_16x16x128_fp8_fp8(a0.v, b2.v, (short)0, acc02, false, false);
        acc03 = __builtin_amdgcn_wmma_f32_16x16x128_fp8_fp8(a0.v, b3.v, (short)0, acc03, false, false);
        acc10 = __builtin_amdgcn_wmma_f32_16x16x128_fp8_fp8(a1.v, b0.v, (short)0, acc10, false, false);
        acc11 = __builtin_amdgcn_wmma_f32_16x16x128_fp8_fp8(a1.v, b1.v, (short)0, acc11, false, false);
        acc12 = __builtin_amdgcn_wmma_f32_16x16x128_fp8_fp8(a1.v, b2.v, (short)0, acc12, false, false);
        acc13 = __builtin_amdgcn_wmma_f32_16x16x128_fp8_fp8(a1.v, b3.v, (short)0, acc13, false, false);

        wait_async0();
        __syncthreads();
    }

    const int c0 = Nblk + Bn0 + lrow;
    const float bv0 = bias[c0];
    const float bv1 = bias[c0 + 16];
    const float bv2 = bias[c0 + 32];
    const float bv3 = bias[c0 + 48];
    const int rb0 = Mblk + Am0 + (half ? 8 : 0);

#pragma unroll
    for (int r = 0; r < 8; ++r) {
        float* orow0 = out + (size_t)(rb0 + r) * DIM_N;
        orow0[c0]      = hw_tanh(acc00[r] + bv0);
        orow0[c0 + 16] = hw_tanh(acc01[r] + bv1);
        orow0[c0 + 32] = hw_tanh(acc02[r] + bv2);
        orow0[c0 + 48] = hw_tanh(acc03[r] + bv3);
        float* orow1 = out + (size_t)(rb0 + 16 + r) * DIM_N;
        orow1[c0]      = hw_tanh(acc10[r] + bv0);
        orow1[c0 + 16] = hw_tanh(acc11[r] + bv1);
        orow1[c0 + 32] = hw_tanh(acc12[r] + bv2);
        orow1[c0 + 48] = hw_tanh(acc13[r] + bv3);
    }
}

extern "C" void kernel_launch(void* const* d_in, const int* in_sizes, int n_in,
                              void* d_out, int out_size, void* d_ws, size_t ws_size,
                              hipStream_t stream) {
    const float* x    = (const float*)d_in[0];
    const float* wgt  = (const float*)d_in[1];
    const float* bias = (const float*)d_in[2];
    float* out = (float*)d_out;

    unsigned char* Abin = (unsigned char*)d_ws;
    unsigned char* Btb  = Abin + (size_t)DIM_B * DIM_K;

    {
        int total_threads = (DIM_B * DIM_K) / 16;
        binarize_x_kernel<<<total_threads / 256, 256, 0, stream>>>(
            x, (unsigned int*)Abin);
    }
    {
        dim3 grid(DIM_N / 64, DIM_K / 64);
        binarize_wT_kernel<<<grid, 256, 0, stream>>>(wgt, Btb);
    }
    {
        dim3 grid(DIM_N / 128, DIM_B / 128);
        bgemm_fp8_kernel<<<grid, 256, 0, stream>>>(Abin, Btb, bias, out);
    }
}
